// GraphFormerBlock_10548439679331
// MI455X (gfx1250) — matrix operand order repaired and checked by reading, not compiled, not run on hardware
//
#include <hip/hip_runtime.h>
#include <hip/hip_bf16.h>
#include <math.h>

// ---------------------------------------------------------------------------
// Types for CDNA5 WMMA (wave32): v_wmma_f32_16x16x32_bf16
// ---------------------------------------------------------------------------
typedef __bf16 bf16;
typedef __bf16 bf16x4 __attribute__((ext_vector_type(4)));
typedef __bf16 bf16x8 __attribute__((ext_vector_type(8)));
typedef __bf16 v16bf  __attribute__((ext_vector_type(16)));
typedef float  v8f    __attribute__((ext_vector_type(8)));
typedef float  f32x4  __attribute__((ext_vector_type(4)));

union BV16 { bf16x8 h[2]; v16bf v; };

#define D_FEAT 128

__device__ __forceinline__ float wave_sum(float s) {
#pragma unroll
    for (int off = 16; off > 0; off >>= 1) s += __shfl_xor(s, off, 32);
    return s;
}

__device__ __forceinline__ void atomicMaxFloat(float* addr, float val) {
    // int-compare trick: works across mixed signs with -inf init
    if (val >= 0.0f) atomicMax((int*)addr, __float_as_int(val));
    else             atomicMin((unsigned int*)addr, __float_as_uint(val));
}

// ---------------------------------------------------------------------------
// WMMA GEMM:  Out[M, NOUT] = A[M, K] * W[NOUT, K]^T + bias
//   Block = 128 threads (4 waves). Column tile = 64 (16 per wave), held in
//   LDS for 8 consecutive 16-row tiles (128 rows) to amortize B staging.
//   EPI: 0 = store bf16, 1 = exact GELU then store bf16, 2 = store f32
//   ABF16: A is bf16 (else f32, converted while staging to LDS)
// ---------------------------------------------------------------------------
template <int K, int NOUT, int EPI, bool ABF16>
__global__ void __launch_bounds__(128)
gemm_wmma(const void* __restrict__ Aram, const float* __restrict__ W,
          const float* __restrict__ bias, void* __restrict__ Outp, int M) {
    constexpr int RT = 8;                       // row tiles per block
    __shared__ bf16 sB[64 * K];
    __shared__ bf16 sA[16 * K];

    const int tid     = threadIdx.x;
    const int cbase   = blockIdx.y * 64;
    const int rowBase = blockIdx.x * (16 * RT);

    // Stage B tile (64 rows of W, K wide) once; f32 -> bf16, float4 loads
    for (int i = tid * 4; i < 64 * K; i += 128 * 4) {
        int r = i / K, c = i % K;
        f32x4 v = *(const f32x4*)(W + (size_t)(cbase + r) * K + c);
        bf16x4 o; o[0] = (bf16)v[0]; o[1] = (bf16)v[1];
                  o[2] = (bf16)v[2]; o[3] = (bf16)v[3];
        *(bf16x4*)(sB + i) = o;
    }

    const int w    = tid >> 5;
    const int lane = tid & 31;
    const int lrow = lane & 15;      // M-row (A) / N-col (B,C) owned by lane
    const int lhi  = lane >> 4;      // lane-half selector
    const int col  = cbase + w * 16 + lrow;
    const float bb = bias[col];

    for (int rt = 0; rt < RT; ++rt) {
        const int row0 = rowBase + rt * 16;
        if (row0 >= M) break;
        __syncthreads();   // B visible (iter 0); sA readers done (iter > 0)

        // Stage A tile (16 x K) into LDS as bf16
        if (ABF16) {
            const bf16* A = (const bf16*)Aram;
            for (int i = tid * 8; i < 16 * K; i += 128 * 8) {
                int r = i / K, c = i % K;
                int gr = row0 + r; if (gr >= M) gr = M - 1;
                *(bf16x8*)(sA + i) = *(const bf16x8*)(A + (size_t)gr * K + c);
            }
        } else {
            const float* A = (const float*)Aram;
            for (int i = tid * 4; i < 16 * K; i += 128 * 4) {
                int r = i / K, c = i % K;
                int gr = row0 + r; if (gr >= M) gr = M - 1;
                f32x4 v = *(const f32x4*)(A + (size_t)gr * K + c);
                bf16x4 o; o[0] = (bf16)v[0]; o[1] = (bf16)v[1];
                          o[2] = (bf16)v[2]; o[3] = (bf16)v[3];
                *(bf16x4*)(sA + i) = o;
            }
        }
        __syncthreads();

        v8f acc = {};
#pragma unroll
        for (int kk = 0; kk < K; kk += 32) {
            // A fragment (16-bit A 16x32 layout)
            int ka = kk + (lhi << 3);
            BV16 a;
            a.h[0] = *(const bf16x8*)(sA + lrow * K + ka);        // K=ka+0..7
            a.h[1] = *(const bf16x8*)(sA + lrow * K + ka + 16);   // K=ka+16..23
            // B fragment (16-bit B 32x16 layout)
            int kb = kk + (lhi << 3);
            const bf16* bp = sB + (size_t)(w * 16 + lrow) * K + kb;
            BV16 b;
            b.h[0] = *(const bf16x8*)(bp);
            b.h[1] = *(const bf16x8*)(bp + 16);
            acc = __builtin_amdgcn_wmma_f32_16x16x32_bf16(
                false, a.v, false, b.v, (short)0, acc, false, false);
        }

        // C/D layout: VGPR v -> M = v + 8*lhi, N = lane&15
        if (row0 + 16 <= M) {
            // Full tile: predicate-free, immediate-offset stores
            if (EPI == 2) {
                float* O = (float*)Outp + (size_t)(row0 + (lhi << 3)) * NOUT + col;
#pragma unroll
                for (int v = 0; v < 8; ++v)
                    O[(size_t)v * NOUT] = acc[v] + bb;
            } else {
                bf16* O = (bf16*)Outp + (size_t)(row0 + (lhi << 3)) * NOUT + col;
#pragma unroll
                for (int v = 0; v < 8; ++v) {
                    float val = acc[v] + bb;
                    if (EPI == 1) val = 0.5f * val * (1.0f + erff(val * 0.70710678118f));
                    O[(size_t)v * NOUT] = (bf16)val;
                }
            }
        } else {
#pragma unroll
            for (int v = 0; v < 8; ++v) {
                int grow = row0 + v + (lhi << 3);
                if (grow < M) {
                    float val = acc[v] + bb;
                    if (EPI == 1) val = 0.5f * val * (1.0f + erff(val * 0.70710678118f));
                    if (EPI == 2) ((float*)Outp)[(size_t)grow * NOUT + col] = val;
                    else          ((bf16*)Outp)[(size_t)grow * NOUT + col] = (bf16)val;
                }
            }
        }
    }
}

// ---------------------------------------------------------------------------
// Init: emax = -inf, denom = 0, hout = 0
// ---------------------------------------------------------------------------
__global__ void init_buffers(float* emax, float* denom, float* hout, int N) {
    int i = blockIdx.x * blockDim.x + threadIdx.x;
    if (i < N * 4) { emax[i] = -__builtin_inff(); denom[i] = 0.0f; }
    if (i < N * D_FEAT) hout[i] = 0.0f;
}

// ---------------------------------------------------------------------------
// Edge scores: one wave per edge (incl. self-loops at index >= E).
// Each lane owns contiguous d = lane*4..lane*4+3 -> 8B bf16x4 gathers.
// e[h] = sum_d att[h,d] * leaky_relu(xl[src,h,d] + xr[dst,h,d])
// ---------------------------------------------------------------------------
__global__ void __launch_bounds__(256)
edge_scores(const bf16* __restrict__ xl, const bf16* __restrict__ xr,
            const int* __restrict__ ei, const float* __restrict__ att,
            float* __restrict__ escore, float* __restrict__ emax,
            int E, int N) {
    int wid  = (blockIdx.x * blockDim.x + threadIdx.x) >> 5;
    int lane = threadIdx.x & 31;
    int total = E + N;
    if (wid >= total) return;
    int src, dst;
    if (wid < E) { src = ei[wid]; dst = ei[E + wid]; }
    else         { src = dst = wid - E; }
    const bf16* pl = xl + (size_t)src * 512;
    const bf16* pr = xr + (size_t)dst * 512;
    const int d0 = lane * 4;
#pragma unroll
    for (int h = 0; h < 4; ++h) {
        bf16x4 vl = *(const bf16x4*)(pl + h * D_FEAT + d0);
        bf16x4 vr = *(const bf16x4*)(pr + h * D_FEAT + d0);
        f32x4  va = *(const f32x4*)(att + h * D_FEAT + d0);
        float s = 0.0f;
#pragma unroll
        for (int i = 0; i < 4; ++i) {
            float v = (float)vl[i] + (float)vr[i];
            v = (v > 0.0f) ? v : 0.2f * v;   // leaky_relu, NEG_SLOPE=0.2
            s += v * va[i];
        }
        s = wave_sum(s);
        if (lane == 0) {
            escore[(size_t)wid * 4 + h] = s;
            atomicMaxFloat(&emax[dst * 4 + h], s);
        }
    }
}

// ---------------------------------------------------------------------------
// ee = exp(e - emax[dst]); denom[dst] += ee   (one thread per edge*head)
// ---------------------------------------------------------------------------
__global__ void edge_exp(const int* __restrict__ ei, float* __restrict__ escore,
                         const float* __restrict__ emax, float* __restrict__ denom,
                         int E, int N) {
    int idx = blockIdx.x * blockDim.x + threadIdx.x;
    int total = (E + N) * 4;
    if (idx >= total) return;
    int i = idx >> 2, h = idx & 3;
    int dst = (i < E) ? ei[E + i] : (i - E);
    float ee = __expf(escore[idx] - emax[dst * 4 + h]);
    escore[idx] = ee;
    atomicAdd(&denom[dst * 4 + h], ee);
}

// ---------------------------------------------------------------------------
// hout[dst,d] += (1/H) * sum_h alpha[h] * xl[src,h,d]  (one wave per edge)
// ---------------------------------------------------------------------------
__global__ void __launch_bounds__(256)
edge_aggregate(const bf16* __restrict__ xl, const int* __restrict__ ei,
               const float* __restrict__ ee, const float* __restrict__ denom,
               float* __restrict__ hout, int E, int N) {
    int wid  = (blockIdx.x * blockDim.x + threadIdx.x) >> 5;
    int lane = threadIdx.x & 31;
    int total = E + N;
    if (wid >= total) return;
    int src, dst;
    if (wid < E) { src = ei[wid]; dst = ei[E + wid]; }
    else         { src = dst = wid - E; }
    float alpha[4];
#pragma unroll
    for (int h = 0; h < 4; ++h)
        alpha[h] = ee[(size_t)wid * 4 + h] / (denom[dst * 4 + h] + 1e-16f);
    const bf16* pl = xl + (size_t)src * 512;
    const int d0 = lane * 4;
    float c[4] = {0.0f, 0.0f, 0.0f, 0.0f};
#pragma unroll
    for (int h = 0; h < 4; ++h) {
        bf16x4 vl = *(const bf16x4*)(pl + h * D_FEAT + d0);
#pragma unroll
        for (int i = 0; i < 4; ++i) c[i] += alpha[h] * (float)vl[i];
    }
    float* hp = hout + (size_t)dst * D_FEAT + d0;
#pragma unroll
    for (int i = 0; i < 4; ++i) atomicAdd(hp + i, 0.25f * c[i]);
}

// ---------------------------------------------------------------------------
// LayerNorm over D=128: out = LN(a + b (+ addvec)) * g + be ; one wave per row
// Lane owns contiguous d = lane*4..+3 -> float4 traffic throughout.
// ---------------------------------------------------------------------------
__global__ void __launch_bounds__(256)
ln_fused(const float* __restrict__ a, const float* __restrict__ b,
         const float* __restrict__ addvec, const float* __restrict__ g,
         const float* __restrict__ be, float* __restrict__ out, int N) {
    int row  = (blockIdx.x * blockDim.x + threadIdx.x) >> 5;
    int lane = threadIdx.x & 31;
    if (row >= N) return;
    const int d0 = lane * 4;
    f32x4 va = *(const f32x4*)(a + (size_t)row * D_FEAT + d0);
    f32x4 vb = *(const f32x4*)(b + (size_t)row * D_FEAT + d0);
    f32x4 v;
    float sum = 0.0f;
#pragma unroll
    for (int i = 0; i < 4; ++i) { v[i] = va[i] + vb[i]; }
    if (addvec) {
        f32x4 vc = *(const f32x4*)(addvec + d0);
#pragma unroll
        for (int i = 0; i < 4; ++i) v[i] += vc[i];
    }
#pragma unroll
    for (int i = 0; i < 4; ++i) sum += v[i];
    float mean = wave_sum(sum) * (1.0f / D_FEAT);
    float var = 0.0f;
#pragma unroll
    for (int i = 0; i < 4; ++i) { float t = v[i] - mean; var += t * t; }
    var = wave_sum(var) * (1.0f / D_FEAT);
    float r = rsqrtf(var + 1e-5f);
    f32x4 vg = *(const f32x4*)(g + d0);
    f32x4 vbe = *(const f32x4*)(be + d0);
    f32x4 o;
#pragma unroll
    for (int i = 0; i < 4; ++i) o[i] = (v[i] - mean) * r * vg[i] + vbe[i];
    *(f32x4*)(out + (size_t)row * D_FEAT + d0) = o;
}

// ---------------------------------------------------------------------------
// Launch
// ---------------------------------------------------------------------------
extern "C" void kernel_launch(void* const* d_in, const int* in_sizes, int n_in,
                              void* d_out, int out_size, void* d_ws, size_t ws_size,
                              hipStream_t stream) {
    const float* x    = (const float*)d_in[0];
    const int*   ei   = (const int*)  d_in[1];
    const float* Wl   = (const float*)d_in[2];
    const float* bl   = (const float*)d_in[3];
    const float* Wr   = (const float*)d_in[4];
    const float* br   = (const float*)d_in[5];
    const float* att  = (const float*)d_in[6];
    const float* gbias= (const float*)d_in[7];
    const float* g1   = (const float*)d_in[8];
    const float* be1  = (const float*)d_in[9];
    const float* W1   = (const float*)d_in[10];
    const float* b1   = (const float*)d_in[11];
    const float* W2   = (const float*)d_in[12];
    const float* b2   = (const float*)d_in[13];
    const float* g2   = (const float*)d_in[14];
    const float* be2  = (const float*)d_in[15];
    float* out = (float*)d_out;

    const int N = in_sizes[0] / D_FEAT;      // 50000
    const int E = in_sizes[1] / 2;           // 400000
    const int T = E + N;                     // edges + self-loops

    // Workspace carve-up
    char* p = (char*)d_ws;
    bf16*  xl    = (bf16*)p;  p += (size_t)N * 512 * sizeof(bf16);   // 51.2 MB (L2-resident)
    bf16*  xr    = (bf16*)p;  p += (size_t)N * 512 * sizeof(bf16);   // 51.2 MB
    float* esc   = (float*)p; p += (size_t)T * 4 * sizeof(float);    // 7.2 MB
    float* emax  = (float*)p; p += (size_t)N * 4 * sizeof(float);
    float* denom = (float*)p; p += (size_t)N * 4 * sizeof(float);
    float* hout  = (float*)p; p += (size_t)N * D_FEAT * sizeof(float);
    float* y1    = (float*)p; p += (size_t)N * D_FEAT * sizeof(float);
    bf16*  mid   = (bf16*)p;  p += (size_t)N * 256 * sizeof(bf16);
    float* h2    = (float*)p; p += (size_t)N * D_FEAT * sizeof(float);

    const int rowBlocks = (N + 127) / 128;   // 128 rows per block

    // Init accumulators (stream-serialized before edge kernels)
    init_buffers<<<(N * D_FEAT + 255) / 256, 256, 0, stream>>>(emax, denom, hout, N);

    // GATv2 linear transforms (bf16 WMMA, f32 accumulate) -> bf16 in L2
    gemm_wmma<128, 512, 0, false><<<dim3(rowBlocks, 8), 128, 0, stream>>>(x, Wl, bl, xl, N);
    gemm_wmma<128, 512, 0, false><<<dim3(rowBlocks, 8), 128, 0, stream>>>(x, Wr, br, xr, N);

    // Attention: score -> softmax -> aggregate
    edge_scores   <<<(T + 7) / 8, 256, 0, stream>>>(xl, xr, ei, att, esc, emax, E, N);
    edge_exp      <<<(T * 4 + 255) / 256, 256, 0, stream>>>(ei, esc, emax, denom, E, N);
    edge_aggregate<<<(T + 7) / 8, 256, 0, stream>>>(xl, ei, esc, denom, hout, E, N);

    // Residual + LN1
    ln_fused<<<(N + 7) / 8, 256, 0, stream>>>(x, hout, gbias, g1, be1, y1, N);

    // FFN: y1 @ W1^T + b1 -> GELU -> bf16 mid ; mid @ W2^T + b2 -> f32 h2
    gemm_wmma<128, 256, 1, false><<<dim3(rowBlocks, 4), 128, 0, stream>>>(y1, W1, b1, mid, N);
    gemm_wmma<256, 128, 2, true ><<<dim3(rowBlocks, 2), 128, 0, stream>>>(mid, W2, b2, h2, N);

    // Residual + LN2 -> output
    ln_fused<<<(N + 7) / 8, 256, 0, stream>>>(y1, h2, nullptr, g2, be2, out, N);
}
